// WorkingRingAttention_76312978915790
// MI455X (gfx1250) — matrix operand order repaired and checked by reading, not compiled, not run on hardware
//
#include <hip/hip_runtime.h>
#include <cstdint>

typedef __attribute__((ext_vector_type(16))) _Float16 v16h;
typedef __attribute__((ext_vector_type(8)))  float    v8f;

#define B_  2
#define S_  2048
#define H_  16
#define D_  64
#define BQ  64          // query rows per workgroup (16 per wave)
#define BK  64          // keys per iteration
#define NW  4           // waves per workgroup
#define PAD 8           // halves of padding: row stride 72 halves = 144B (16B multiple)

// Issue one tile (64 rows x 64 f32) as 8 async b128 loads per thread-slice.
// 128 threads x 8 chunks x 16B = 16KB = full tile, lands in LDS stage buffer.
__device__ __forceinline__ void async_stage_tile(const float* gbase, int k0,
                                                 unsigned ldsOff, int tid,
                                                 size_t strideS)
{
    #pragma unroll
    for (int j = 0; j < 8; ++j) {
        const int cid = j * 128 + tid;       // 0..1023 unique chunk id
        const int row = cid >> 4;            // key row 0..63
        const int seg = cid & 15;            // 16B segment within row
        const float*   gp = gbase + (size_t)(k0 + row) * strideS + seg * 4;
        const uint64_t ga = (uint64_t)(uintptr_t)gp;
        const unsigned la = ldsOff + (unsigned)(row * 256 + seg * 16);
        asm volatile("global_load_async_to_lds_b128 %0, %1, off"
                     :: "v"(la), "v"(ga) : "memory");
    }
}

// Flash attention, one (b,h,qtile) per workgroup, wave32, WMMA f16->f32,
// async-engine double-buffered K/V staging.
__global__ __launch_bounds__(128) void fa_wmma_kernel(
    const float* __restrict__ qg, const float* __restrict__ kg,
    const float* __restrict__ vg, float* __restrict__ og)
{
    __shared__ float    SgK[BK][D_];              // f32 stage (async dst)
    __shared__ float    SgV[BK][D_];              // f32 stage (async dst)
    __shared__ _Float16 Kt [BK][D_ + PAD];        // K tile f16: [key][d]
    __shared__ _Float16 Vt [D_][BK + PAD];        // V tile f16 transposed: [d][key]
    __shared__ _Float16 Pst[NW][16][BK + PAD];    // per-wave P staging (C/D -> A)

    const int tid  = threadIdx.x;
    const int wave = tid >> 5;
    const int lane = tid & 31;
    const int ln   = lane & 15;     // column / row-in-half index
    const int hi   = lane >> 4;     // lane half (0/1)

    const int blk = blockIdx.x;                 // 0 .. B*H*(S/BQ)-1
    const int qt  = blk % (S_ / BQ);
    const int bh  = blk / (S_ / BQ);
    const int h   = bh % H_;
    const int b   = bh / H_;

    const size_t strideS = (size_t)H_ * D_;     // floats between consecutive s
    const int    q0      = qt * BQ + wave * 16; // first query row of this wave

    const float* qbase = qg + (((size_t)b * S_ + q0) * H_ + h) * D_;
    const float* kbase = kg + (((size_t)b * S_) * H_ + h) * D_;
    const float* vbase = vg + (((size_t)b * S_) * H_ + h) * D_;
    float*       obase = og + (((size_t)b * S_ + q0) * H_ + h) * D_;

    const unsigned skOff = (unsigned)(uintptr_t)&SgK[0][0];  // LDS byte offsets
    const unsigned svOff = (unsigned)(uintptr_t)&SgV[0][0];

    // ---------------- Q -> A fragments (16x32 f16 each), folded 1/sqrt(D) ----------------
    // A layout (16-bit, 16x32): lane row = ln; halves hold 16 contiguous
    // d = 32c + b0 + {0..15}, with b0 = hi ? 16 : 0.
    v16h Aq[2];
    {
        const float  sc = 0.125f;               // 1/sqrt(64)
        const float* qr = qbase + (size_t)ln * strideS;
        #pragma unroll
        for (int c = 0; c < 2; ++c) {
            const int d0 = 32 * c + (hi ? 16 : 0);
            #pragma unroll
            for (int j = 0; j < 8; ++j) {
                Aq[c][j]     = (_Float16)(qr[d0 + j]      * sc);
                Aq[c][j + 8] = (_Float16)(qr[d0 + 8 + j] * sc);
            }
        }
    }

    v8f   Oacc[4] = {};                         // 16x64 f32 output accum (4 N-tiles)
    float mrow[8], lrow[8];
    #pragma unroll
    for (int r = 0; r < 8; ++r) { mrow[r] = -1e30f; lrow[r] = 0.0f; }

    // ---- preload tile 0 through the async engine
    async_stage_tile(kbase, 0, skOff, tid, strideS);
    async_stage_tile(vbase, 0, svOff, tid, strideS);

    const int NIT = S_ / BK;
    for (int it = 0; it < NIT; ++it) {
        // my async writes done; barrier also fences prev-iter readers of Kt/Vt
        asm volatile("s_wait_asynccnt 0x0" ::: "memory");
        __syncthreads();

        // -------- convert staged f32 -> f16 tiles (once per workgroup) --------
        {
            const int row = tid >> 1;            // key 0..63
            const int dh  = (tid & 1) * 32;      // 0 or 32
            const float* sk = &SgK[row][dh];
            const float* sv = &SgV[row][dh];
            #pragma unroll
            for (int j = 0; j < 32; ++j)
                Kt[row][dh + j] = (_Float16)sk[j];   // [key][d]: d-contiguous
            #pragma unroll
            for (int j = 0; j < 32; ++j)
                Vt[dh + j][row] = (_Float16)sv[j];   // transpose: [d][key]
        }
        __syncthreads();   // Kt/Vt ready; stage buffers free for reuse

        // -------- kick off next tile on the async engine, overlapped with compute
        if (it + 1 < NIT) {
            const int kn = (it + 1) * BK;
            async_stage_tile(kbase, kn, skOff, tid, strideS);
            async_stage_tile(vbase, kn, svOff, tid, strideS);
        }

        // -------- S = Q K^T : 4 key N-tiles x 2 d-chunks of 32 --------
        // B layout (16-bit, 32x16): lane col n = ln; halves hold 16 contiguous
        // d values starting at 32c + (hi ? 16 : 0) -> read from Kt[key][d].
        v8f Sacc[4];
        #pragma unroll
        for (int t = 0; t < 4; ++t) {
            v8f cc = {};
            #pragma unroll
            for (int c = 0; c < 2; ++c) {
                const _Float16* kr = &Kt[t * 16 + ln][32 * c + (hi ? 16 : 0)];
                v16h Bk;
                #pragma unroll
                for (int j = 0; j < 16; ++j) Bk[j] = kr[j];
                cc = __builtin_amdgcn_wmma_f32_16x16x32_f16(
                        false, Aq[c], false, Bk, (short)0, cc, false, false);
            }
            Sacc[t] = cc;
        }

        // -------- online softmax on C/D layout --------
        // VGPR r holds row (hi*8 + r), column = t*16 + ln across the 4 tiles.
        #pragma unroll
        for (int r = 0; r < 8; ++r) {
            float mx = Sacc[0][r];
            mx = fmaxf(mx, Sacc[1][r]);
            mx = fmaxf(mx, Sacc[2][r]);
            mx = fmaxf(mx, Sacc[3][r]);
            #pragma unroll
            for (int off = 1; off < 16; off <<= 1)
                mx = fmaxf(mx, __shfl_xor(mx, off, 32));

            const float mnew = fmaxf(mrow[r], mx);
            const float corr = __expf(mrow[r] - mnew);
            mrow[r] = mnew;

            float rs = 0.0f;
            #pragma unroll
            for (int t = 0; t < 4; ++t) {
                const float p = __expf(Sacc[t][r] - mnew);
                Sacc[t][r] = p;
                rs += p;
            }
            #pragma unroll
            for (int off = 1; off < 16; off <<= 1)
                rs += __shfl_xor(rs, off, 32);

            lrow[r] = lrow[r] * corr + rs;
            #pragma unroll
            for (int t = 0; t < 4; ++t) Oacc[t][r] *= corr;
        }

        // -------- stage P (f16) through LDS to re-layout C/D -> A --------
        #pragma unroll
        for (int r = 0; r < 8; ++r) {
            const int row = hi * 8 + r;
            #pragma unroll
            for (int t = 0; t < 4; ++t)
                Pst[wave][row][t * 16 + ln] = (_Float16)Sacc[t][r];
        }
        // producer == consumer wave: only need the DS counter drained, no barrier
        asm volatile("s_wait_dscnt 0x0" ::: "memory");

        // -------- O += P V : 4 d N-tiles x 2 key-chunks of 32 --------
        #pragma unroll
        for (int c = 0; c < 2; ++c) {
            // A fragment of P: lane row = ln; halves = key 32c + b0 + {0..15}
            v16h Pa;
            const int kb = 32 * c + (hi ? 16 : 0);
            #pragma unroll
            for (int j = 0; j < 8; ++j) {
                Pa[j]     = Pst[wave][ln][kb + j];
                Pa[j + 8] = Pst[wave][ln][kb + 8 + j];
            }
            #pragma unroll
            for (int t = 0; t < 4; ++t) {
                // B fragment of V: lane col d = t*16 + ln; halves = 16 contiguous keys
                v16h Vb;
                const _Float16* vr = &Vt[t * 16 + ln][32 * c + (hi ? 16 : 0)];
                #pragma unroll
                for (int j = 0; j < 16; ++j) Vb[j] = vr[j];
                Oacc[t] = __builtin_amdgcn_wmma_f32_16x16x32_f16(
                            false, Pa, false, Vb, (short)0, Oacc[t], false, false);
            }
        }
    }

    // ---------------- epilogue: O / l, store fp32 ----------------
    #pragma unroll
    for (int r = 0; r < 8; ++r) {
        const float inv = __frcp_rn(lrow[r]);
        const int   row = hi * 8 + r;
        float* orow = obase + (size_t)row * strideS;
        #pragma unroll
        for (int t = 0; t < 4; ++t)
            orow[t * 16 + ln] = Oacc[t][r] * inv;
    }
}

extern "C" void kernel_launch(void* const* d_in, const int* in_sizes, int n_in,
                              void* d_out, int out_size, void* d_ws, size_t ws_size,
                              hipStream_t stream) {
    const float* q = (const float*)d_in[0];
    const float* k = (const float*)d_in[1];
    const float* v = (const float*)d_in[2];
    float* out = (float*)d_out;

    dim3 grid(B_ * H_ * (S_ / BQ));   // 1024 workgroups
    dim3 block(128);                  // 4 waves (wave32)
    hipLaunchKernelGGL(fa_wmma_kernel, grid, block, 0, stream, q, k, v, out);
}
